// TransformerBlock_55387898249509
// MI455X (gfx1250) — matrix operand order repaired and checked by reading, not compiled, not run on hardware
//
#include <hip/hip_runtime.h>
#include <stdint.h>

#define DEV __device__ __forceinline__

typedef __attribute__((ext_vector_type(16))) __bf16 v16bf;
typedef __attribute__((ext_vector_type(8)))  float  v8f;

static constexpr int kB = 4, kS = 2048, kD = 2048, kH = 16, kHD = 128, kFF = 5632;
static constexpr int kM = kB * kS;

// ---------------- CDNA5 Tensor Data Mover support (compile-guarded) ----------
#if defined(__AMDGCN__) && defined(__has_builtin)
#if __has_builtin(__builtin_amdgcn_tensor_load_to_lds) && __has_builtin(__builtin_amdgcn_s_wait_tensorcnt)
#define USE_TDM 1
#else
#define USE_TDM 0
#endif
#else
#define USE_TDM 0
#endif

#if USE_TDM
#if __has_include(<hip/amd_detail/amd_gfx1250_TDM.h>)
#define TDM_ARGS6 1
#else
#define TDM_ARGS6 0
#endif

typedef __attribute__((ext_vector_type(4))) unsigned int v4u;
typedef __attribute__((ext_vector_type(4))) int v4i;
typedef __attribute__((ext_vector_type(8))) int v8i;

DEV unsigned int lds_off(const void* p) {
    // flat shared pointer: low 32 bits are the LDS byte address (ISA 10.2 aperture rules)
    return (unsigned int)(uintptr_t)p;
}

// Issue a TDM load of a 2-D bf16 tile [tile_d1 rows x tile_d0 elems] from a
// row-major tensor with row stride stride0 (elements) into contiguous LDS.
DEV void tdm_load_tile_bf16(unsigned int lds_addr, const unsigned short* gptr,
                            unsigned int tensor_d0, unsigned int tensor_d1,
                            unsigned int tile_d0, unsigned int tile_d1,
                            unsigned long long stride0) {
    unsigned long long ga = (unsigned long long)(uintptr_t)gptr;
    v4u g0;
    g0[0] = 1u;                                   // count=1, user descriptor
    g0[1] = lds_addr;                             // LDS byte address
    g0[2] = (unsigned int)(ga & 0xFFFFFFFFu);     // global_addr[31:0]
    g0[3] = (unsigned int)((ga >> 32) & 0x01FFFFFFu) | (2u << 30);  // addr[56:32], type=2
    v8i g1;
    g1[0] = (int)(1u << 16);                                      // data_size=1 (2 bytes)
    g1[1] = (int)((tensor_d0 & 0xFFFFu) << 16);                   // tensor_dim0 lo16
    g1[2] = (int)((tensor_d0 >> 16) | ((tensor_d1 & 0xFFFFu) << 16));
    g1[3] = (int)((tensor_d1 >> 16) | (tile_d0 << 16));           // tile_dim0
    g1[4] = (int)(tile_d1 & 0xFFFFu);                             // tile_dim1 (tile_dim2=0)
    g1[5] = (int)(stride0 & 0xFFFFFFFFu);                         // dim0 stride lo32
    g1[6] = (int)((stride0 >> 32) & 0xFFFFu);                     // dim0 stride hi16
    g1[7] = 0;
    v4i z4; z4[0] = z4[1] = z4[2] = z4[3] = 0;
#if TDM_ARGS6
    v8i z8;
    #pragma unroll
    for (int i = 0; i < 8; ++i) z8[i] = 0;
    __builtin_amdgcn_tensor_load_to_lds(g0, g1, z4, z4, z8, 0);
#else
    __builtin_amdgcn_tensor_load_to_lds(g0, g1, z4, z4, 0);
#endif
}
#endif  // USE_TDM

// ---------------------------------------------------------------- helpers
union Frag {
    uint4 q[2];
    v16bf v;
};

DEV unsigned short f2bf(float f) {
    union { float f; uint32_t u; } c; c.f = f;
    uint32_t u = c.u;
    return (unsigned short)((u + 0x7FFFu + ((u >> 16) & 1u)) >> 16);
}
DEV float bf2f(unsigned short h) {
    union { uint32_t u; float f; } c; c.u = ((uint32_t)h) << 16;
    return c.f;
}
DEV v8f vzero() {
    v8f z;
    #pragma unroll
    for (int i = 0; i < 8; ++i) z[i] = 0.0f;
    return z;
}
DEV v8f bf16_wmma(const Frag& a, const Frag& b, v8f c) {
    return __builtin_amdgcn_wmma_f32_16x16x32_bf16(false, a.v, false, b.v,
                                                   (short)0, c, false, false);
}

// ---------------------------------------------------------------- cvt f32->bf16
__global__ __launch_bounds__(256) void k_cvt_bf16(const float* __restrict__ src,
                                                  unsigned short* __restrict__ dst,
                                                  int n) {
    int i = blockIdx.x * 256 + threadIdx.x;
    if (i < n) dst[i] = f2bf(src[i]);
}

// ---------------------------------------------------------------- RMSNorm -> bf16
__global__ __launch_bounds__(256) void k_rmsnorm_bf16(const float* __restrict__ x,
                                                      const float* __restrict__ w,
                                                      unsigned short* __restrict__ out) {
    __shared__ float red[256];
    const int row = blockIdx.x;
    const int tid = threadIdx.x;
    const float* xr = x + (size_t)row * kD;
    float ss = 0.0f;
    #pragma unroll
    for (int i = 0; i < kD / 256; ++i) {
        float v = xr[tid + i * 256];
        ss += v * v;
    }
    red[tid] = ss;
    __syncthreads();
    for (int s = 128; s > 0; s >>= 1) {
        if (tid < s) red[tid] += red[tid + s];
        __syncthreads();
    }
    const float inv = rsqrtf(red[0] / (float)kD + 1e-5f);
    unsigned short* orow = out + (size_t)row * kD;
    #pragma unroll
    for (int i = 0; i < kD / 256; ++i) {
        int c = tid + i * 256;
        orow[c] = f2bf(xr[c] * inv * w[c]);
    }
}

// ---------------------------------------------------------------- RoPE in-place on bf16 q,k
__global__ __launch_bounds__(256) void k_rope_inplace(unsigned short* __restrict__ q,
                                                      unsigned short* __restrict__ k,
                                                      const float* __restrict__ cosT,
                                                      const float* __restrict__ sinT) {
    int idx = blockIdx.x * 256 + threadIdx.x;   // [0, B*S*H*HD/2)
    int i  = idx & 63;
    int h  = (idx >> 6) & 15;
    int s  = (idx >> 10) & 2047;
    int b  = idx >> 21;
    size_t base = (((size_t)b * kS + s) * kD) + (size_t)h * kHD + 2 * i;
    float c  = cosT[s * 64 + i];
    float sn = sinT[s * 64 + i];
    float q0 = bf2f(q[base]), q1 = bf2f(q[base + 1]);
    q[base]     = f2bf(q0 * c - q1 * sn);
    q[base + 1] = f2bf(q0 * sn + q1 * c);
    float k0 = bf2f(k[base]), k1 = bf2f(k[base + 1]);
    k[base]     = f2bf(k0 * c - k1 * sn);
    k[base + 1] = f2bf(k0 * sn + k1 * c);
}

// ---------------------------------------------------------------- NT GEMM via WMMA
// C[m,n] = sum_k A[m,k] * B[n,k];  A:[M,K] bf16 row-major, B:[N,K] bf16 row-major.
constexpr int GM_BF16 = 0;   // store bf16
constexpr int GM_RESID = 1;  // outF = acc + resid (f32)
constexpr int GM_ADD = 2;    // outF += acc (f32)

template <int MODE>
__global__ __launch_bounds__(256) void k_gemm_nt(
    const unsigned short* __restrict__ A,
    const unsigned short* __restrict__ Bm,
    const float* __restrict__ resid,
    unsigned short* __restrict__ outB,
    float* __restrict__ outF,
    int M, int N, int K)
{
    const int tid  = threadIdx.x;
    const int wave = tid >> 5;
    const int lane = tid & 31;
    const int hlf  = lane >> 4;
    const int lm   = lane & 15;
    const int waveM = wave >> 2;   // 0..1 -> 64 rows each
    const int waveN = wave & 3;    // 0..3 -> 32 cols each
    const int mBase = blockIdx.y * 128;
    const int nBase = blockIdx.x * 128;

    v8f acc[4][2];
    #pragma unroll
    for (int i = 0; i < 4; ++i)
        #pragma unroll
        for (int j = 0; j < 2; ++j) acc[i][j] = vzero();

#if USE_TDM
    // ---- TDM double-buffered pipeline: wave 0 drives the Tensor Data Mover ----
    __shared__ __align__(16) unsigned short As[2][128][32];
    __shared__ __align__(16) unsigned short Bs[2][128][32];
    const bool issuer = (wave == 0);
    if (issuer) {
        tdm_load_tile_bf16(lds_off(&As[0][0][0]), &A[(size_t)mBase * K], K, 128, 32, 128, K);
        tdm_load_tile_bf16(lds_off(&Bs[0][0][0]), &Bm[(size_t)nBase * K], K, 128, 32, 128, K);
    }
    for (int k0 = 0; k0 < K; k0 += 32) {
        const int p = (k0 >> 5) & 1;
        __syncthreads();  // everyone done reading buffer 1-p
        if (issuer) {
            if (k0 + 32 < K) {
                tdm_load_tile_bf16(lds_off(&As[1 - p][0][0]),
                                   &A[(size_t)mBase * K + k0 + 32], K, 128, 32, 128, K);
                tdm_load_tile_bf16(lds_off(&Bs[1 - p][0][0]),
                                   &Bm[(size_t)nBase * K + k0 + 32], K, 128, 32, 128, K);
                __builtin_amdgcn_s_wait_tensorcnt((short)2);  // buffer p complete (in-order)
            } else {
                __builtin_amdgcn_s_wait_tensorcnt((short)0);
            }
        }
        __syncthreads();  // buffer p published to the workgroup

        Frag af[4], bfr[2];
        #pragma unroll
        for (int mf = 0; mf < 4; ++mf) {
            int r = waveM * 64 + mf * 16 + lm;
            af[mf].q[0] = *(const uint4*)&As[p][r][hlf * 16];
            af[mf].q[1] = *(const uint4*)&As[p][r][hlf * 16 + 8];
        }
        #pragma unroll
        for (int nf = 0; nf < 2; ++nf) {
            int r = waveN * 32 + nf * 16 + lm;
            bfr[nf].q[0] = *(const uint4*)&Bs[p][r][hlf * 16];
            bfr[nf].q[1] = *(const uint4*)&Bs[p][r][hlf * 16 + 8];
        }
        #pragma unroll
        for (int mf = 0; mf < 4; ++mf)
            #pragma unroll
            for (int nf = 0; nf < 2; ++nf)
                acc[mf][nf] = bf16_wmma(af[mf], bfr[nf], acc[mf][nf]);
    }
#else
    // ---- fallback: synchronous vector-load staging ----
    __shared__ __align__(16) unsigned short As[1][128][32];
    __shared__ __align__(16) unsigned short Bs[1][128][32];
    for (int k0 = 0; k0 < K; k0 += 32) {
        __syncthreads();
        #pragma unroll
        for (int t = 0; t < 2; ++t) {
            int vec = tid + t * 256;
            int row = vec >> 2, seg = vec & 3;
            *(uint4*)&As[0][row][seg * 8] =
                *(const uint4*)&A[(size_t)(mBase + row) * K + k0 + seg * 8];
            *(uint4*)&Bs[0][row][seg * 8] =
                *(const uint4*)&Bm[(size_t)(nBase + row) * K + k0 + seg * 8];
        }
        if (k0 + 32 < K) {
            __builtin_prefetch(&A[(size_t)(mBase + (tid >> 1)) * K + k0 + 32], 0, 0);
            __builtin_prefetch(&Bm[(size_t)(nBase + (tid >> 1)) * K + k0 + 32], 0, 0);
        }
        __syncthreads();

        Frag af[4], bfr[2];
        #pragma unroll
        for (int mf = 0; mf < 4; ++mf) {
            int r = waveM * 64 + mf * 16 + lm;
            af[mf].q[0] = *(const uint4*)&As[0][r][hlf * 16];
            af[mf].q[1] = *(const uint4*)&As[0][r][hlf * 16 + 8];
        }
        #pragma unroll
        for (int nf = 0; nf < 2; ++nf) {
            int r = waveN * 32 + nf * 16 + lm;
            bfr[nf].q[0] = *(const uint4*)&Bs[0][r][hlf * 16];
            bfr[nf].q[1] = *(const uint4*)&Bs[0][r][hlf * 16 + 8];
        }
        #pragma unroll
        for (int mf = 0; mf < 4; ++mf)
            #pragma unroll
            for (int nf = 0; nf < 2; ++nf)
                acc[mf][nf] = bf16_wmma(af[mf], bfr[nf], acc[mf][nf]);
    }
#endif

    #pragma unroll
    for (int mf = 0; mf < 4; ++mf)
        #pragma unroll
        for (int nf = 0; nf < 2; ++nf)
            #pragma unroll
            for (int r = 0; r < 8; ++r) {
                int row = mBase + waveM * 64 + mf * 16 + r + hlf * 8;
                int col = nBase + waveN * 32 + nf * 16 + lm;
                float v = acc[mf][nf][r];
                size_t o = (size_t)row * N + col;
                if (MODE == GM_BF16)       outB[o] = f2bf(v);
                else if (MODE == GM_RESID) outF[o] = v + resid[o];
                else                       outF[o] += v;
            }
}

// ---------------------------------------------------------------- fused SwiGLU dual-B GEMM
__global__ __launch_bounds__(256) void k_gemm_swiglu(
    const unsigned short* __restrict__ A,   // [M,K]
    const unsigned short* __restrict__ B1,  // [N,K] gate (w1)
    const unsigned short* __restrict__ B3,  // [N,K] up   (w3)
    unsigned short* __restrict__ outB,      // [M,N] bf16 = silu(g)*u
    int M, int N, int K)
{
    const int tid  = threadIdx.x;
    const int wave = tid >> 5;
    const int lane = tid & 31;
    const int hlf  = lane >> 4;
    const int lm   = lane & 15;
    const int waveM = wave >> 1;   // 0..3 -> 32 rows each
    const int waveN = wave & 1;    // 0..1 -> 32 cols each
    const int mBase = blockIdx.y * 128;
    const int nBase = blockIdx.x * 64;

    v8f acc1[2][2], acc3[2][2];
    #pragma unroll
    for (int i = 0; i < 2; ++i)
        #pragma unroll
        for (int j = 0; j < 2; ++j) { acc1[i][j] = vzero(); acc3[i][j] = vzero(); }

#if USE_TDM
    __shared__ __align__(16) unsigned short As[2][128][32];
    __shared__ __align__(16) unsigned short B1s[2][64][32];
    __shared__ __align__(16) unsigned short B3s[2][64][32];
    const bool issuer = (wave == 0);
    if (issuer) {
        tdm_load_tile_bf16(lds_off(&As[0][0][0]),  &A[(size_t)mBase * K],  K, 128, 32, 128, K);
        tdm_load_tile_bf16(lds_off(&B1s[0][0][0]), &B1[(size_t)nBase * K], K, 64, 32, 64, K);
        tdm_load_tile_bf16(lds_off(&B3s[0][0][0]), &B3[(size_t)nBase * K], K, 64, 32, 64, K);
    }
    for (int k0 = 0; k0 < K; k0 += 32) {
        const int p = (k0 >> 5) & 1;
        __syncthreads();
        if (issuer) {
            if (k0 + 32 < K) {
                tdm_load_tile_bf16(lds_off(&As[1 - p][0][0]),
                                   &A[(size_t)mBase * K + k0 + 32], K, 128, 32, 128, K);
                tdm_load_tile_bf16(lds_off(&B1s[1 - p][0][0]),
                                   &B1[(size_t)nBase * K + k0 + 32], K, 64, 32, 64, K);
                tdm_load_tile_bf16(lds_off(&B3s[1 - p][0][0]),
                                   &B3[(size_t)nBase * K + k0 + 32], K, 64, 32, 64, K);
                __builtin_amdgcn_s_wait_tensorcnt((short)3);
            } else {
                __builtin_amdgcn_s_wait_tensorcnt((short)0);
            }
        }
        __syncthreads();

        Frag af[2], b1f[2], b3f[2];
        #pragma unroll
        for (int mf = 0; mf < 2; ++mf) {
            int r = waveM * 32 + mf * 16 + lm;
            af[mf].q[0] = *(const uint4*)&As[p][r][hlf * 16];
            af[mf].q[1] = *(const uint4*)&As[p][r][hlf * 16 + 8];
        }
        #pragma unroll
        for (int nf = 0; nf < 2; ++nf) {
            int r = waveN * 32 + nf * 16 + lm;
            b1f[nf].q[0] = *(const uint4*)&B1s[p][r][hlf * 16];
            b1f[nf].q[1] = *(const uint4*)&B1s[p][r][hlf * 16 + 8];
            b3f[nf].q[0] = *(const uint4*)&B3s[p][r][hlf * 16];
            b3f[nf].q[1] = *(const uint4*)&B3s[p][r][hlf * 16 + 8];
        }
        #pragma unroll
        for (int mf = 0; mf < 2; ++mf)
            #pragma unroll
            for (int nf = 0; nf < 2; ++nf) {
                acc1[mf][nf] = bf16_wmma(af[mf], b1f[nf], acc1[mf][nf]);
                acc3[mf][nf] = bf16_wmma(af[mf], b3f[nf], acc3[mf][nf]);
            }
    }
#else
    __shared__ __align__(16) unsigned short As[1][128][32];
    __shared__ __align__(16) unsigned short B1s[1][64][32];
    __shared__ __align__(16) unsigned short B3s[1][64][32];
    for (int k0 = 0; k0 < K; k0 += 32) {
        __syncthreads();
        #pragma unroll
        for (int t = 0; t < 2; ++t) {
            int vec = tid + t * 256;
            int row = vec >> 2, seg = vec & 3;
            *(uint4*)&As[0][row][seg * 8] =
                *(const uint4*)&A[(size_t)(mBase + row) * K + k0 + seg * 8];
        }
        {
            int row = tid >> 2, seg = tid & 3;
            *(uint4*)&B1s[0][row][seg * 8] =
                *(const uint4*)&B1[(size_t)(nBase + row) * K + k0 + seg * 8];
            *(uint4*)&B3s[0][row][seg * 8] =
                *(const uint4*)&B3[(size_t)(nBase + row) * K + k0 + seg * 8];
        }
        __syncthreads();

        Frag af[2], b1f[2], b3f[2];
        #pragma unroll
        for (int mf = 0; mf < 2; ++mf) {
            int r = waveM * 32 + mf * 16 + lm;
            af[mf].q[0] = *(const uint4*)&As[0][r][hlf * 16];
            af[mf].q[1] = *(const uint4*)&As[0][r][hlf * 16 + 8];
        }
        #pragma unroll
        for (int nf = 0; nf < 2; ++nf) {
            int r = waveN * 32 + nf * 16 + lm;
            b1f[nf].q[0] = *(const uint4*)&B1s[0][r][hlf * 16];
            b1f[nf].q[1] = *(const uint4*)&B1s[0][r][hlf * 16 + 8];
            b3f[nf].q[0] = *(const uint4*)&B3s[0][r][hlf * 16];
            b3f[nf].q[1] = *(const uint4*)&B3s[0][r][hlf * 16 + 8];
        }
        #pragma unroll
        for (int mf = 0; mf < 2; ++mf)
            #pragma unroll
            for (int nf = 0; nf < 2; ++nf) {
                acc1[mf][nf] = bf16_wmma(af[mf], b1f[nf], acc1[mf][nf]);
                acc3[mf][nf] = bf16_wmma(af[mf], b3f[nf], acc3[mf][nf]);
            }
    }
#endif

    #pragma unroll
    for (int mf = 0; mf < 2; ++mf)
        #pragma unroll
        for (int nf = 0; nf < 2; ++nf)
            #pragma unroll
            for (int r = 0; r < 8; ++r) {
                int row = mBase + waveM * 32 + mf * 16 + r + hlf * 8;
                int col = nBase + waveN * 32 + nf * 16 + lm;
                float g = acc1[mf][nf][r];
                float u = acc3[mf][nf][r];
                float act = (g / (1.0f + __expf(-g))) * u;
                outB[(size_t)row * N + col] = f2bf(act);
            }
}

// ---------------------------------------------------------------- flash attention (causal)
__global__ __launch_bounds__(256) void k_flash_attn(
    const unsigned short* __restrict__ Q,   // [B*S, D] bf16, col = h*HD+d
    const unsigned short* __restrict__ Kb,
    const unsigned short* __restrict__ V,
    unsigned short* __restrict__ O)
{
    __shared__ __align__(16) unsigned short Ks[32][128];    // [kv][d]
    __shared__ __align__(16) unsigned short Vt[128][32];    // [d][kv]
    __shared__ __align__(16) unsigned short Pls[8][16][32]; // per-wave P tile

    const int tid  = threadIdx.x;
    const int wave = tid >> 5;
    const int lane = tid & 31;
    const int hlf  = lane >> 4;
    const int lm   = lane & 15;
    const int b = blockIdx.z, h = blockIdx.y;
    const int qBase = blockIdx.x * 128;

    // Q fragments: 16 rows x HD=128, 4 A-frags of 16x32
    Frag qf[4];
    {
        const unsigned short* qrow =
            Q + ((size_t)(b * kS + qBase + wave * 16 + lm)) * kD + (size_t)h * kHD;
        #pragma unroll
        for (int kk = 0; kk < 4; ++kk) {
            qf[kk].q[0] = *(const uint4*)&qrow[kk * 32 + hlf * 16];
            qf[kk].q[1] = *(const uint4*)&qrow[kk * 32 + hlf * 16 + 8];
        }
    }

    v8f o[8];
    #pragma unroll
    for (int f = 0; f < 8; ++f) o[f] = vzero();
    float mrow[8], lrow[8];
    #pragma unroll
    for (int r = 0; r < 8; ++r) { mrow[r] = -3.0e38f; lrow[r] = 0.0f; }

    const float scale = 0.08838834764831845f;  // 1/sqrt(128)
    const int nChunks = qBase / 32 + 4;

    for (int c = 0; c < nChunks; ++c) {
        const int kvBase = c * 32;
        __syncthreads();
        // stage K chunk [32][128]
        #pragma unroll
        for (int t = 0; t < 2; ++t) {
            int vec = tid + t * 256;
            int row = vec >> 4, seg = vec & 15;
            *(uint4*)&Ks[row][seg * 8] =
                *(const uint4*)&Kb[((size_t)(b * kS + kvBase + row)) * kD + (size_t)h * kHD + seg * 8];
        }
        // stage V chunk transposed -> Vt[d][kv]
        #pragma unroll
        for (int t = 0; t < 8; ++t) {
            int vec = tid + t * 256;  // [0,2048)
            int kr = vec >> 6, dp = vec & 63;
            uint32_t w = *(const uint32_t*)
                &V[((size_t)(b * kS + kvBase + kr)) * kD + (size_t)h * kHD + dp * 2];
            Vt[dp * 2][kr]     = (unsigned short)(w & 0xFFFFu);
            Vt[dp * 2 + 1][kr] = (unsigned short)(w >> 16);
        }
        __syncthreads();

        // scores: two 16x16 frags covering 32 kv columns
        v8f sf[2];
        sf[0] = vzero(); sf[1] = vzero();
        #pragma unroll
        for (int nf = 0; nf < 2; ++nf) {
            int n = nf * 16 + lm;
            #pragma unroll
            for (int kk = 0; kk < 4; ++kk) {
                Frag kf;
                kf.q[0] = *(const uint4*)&Ks[n][kk * 32 + hlf * 16];
                kf.q[1] = *(const uint4*)&Ks[n][kk * 32 + hlf * 16 + 8];
                sf[nf] = bf16_wmma(qf[kk], kf, sf[nf]);
            }
        }

        // causal mask + scale + per-row chunk max
        float chmax[8];
        #pragma unroll
        for (int r = 0; r < 8; ++r) chmax[r] = -3.0e38f;
        #pragma unroll
        for (int nf = 0; nf < 2; ++nf)
            #pragma unroll
            for (int r = 0; r < 8; ++r) {
                float v = sf[nf][r] * scale;
                int kv = kvBase + nf * 16 + lm;
                int qg = qBase + wave * 16 + r + hlf * 8;
                v = (kv > qg) ? -3.0e38f : v;
                sf[nf][r] = v;
                chmax[r] = fmaxf(chmax[r], v);
            }
        #pragma unroll
        for (int mask = 1; mask < 16; mask <<= 1)
            #pragma unroll
            for (int r = 0; r < 8; ++r)
                chmax[r] = fmaxf(chmax[r], __shfl_xor(chmax[r], mask, 32));

        float pscale[8], chsum[8];
        #pragma unroll
        for (int r = 0; r < 8; ++r) {
            float mnew = fmaxf(mrow[r], chmax[r]);
            pscale[r] = __expf(mrow[r] - mnew);
            mrow[r] = mnew;
            chsum[r] = 0.0f;
        }
        #pragma unroll
        for (int nf = 0; nf < 2; ++nf)
            #pragma unroll
            for (int r = 0; r < 8; ++r) {
                float p = __expf(sf[nf][r] - mrow[r]);
                sf[nf][r] = p;
                chsum[r] += p;
            }
        #pragma unroll
        for (int mask = 1; mask < 16; mask <<= 1)
            #pragma unroll
            for (int r = 0; r < 8; ++r)
                chsum[r] += __shfl_xor(chsum[r], mask, 32);
        #pragma unroll
        for (int r = 0; r < 8; ++r)
            lrow[r] = lrow[r] * pscale[r] + chsum[r];
        #pragma unroll
        for (int f = 0; f < 8; ++f)
            #pragma unroll
            for (int r = 0; r < 8; ++r)
                o[f][r] *= pscale[r];

        // re-pack P (C-layout) -> A-layout through wave-private LDS
        #pragma unroll
        for (int nf = 0; nf < 2; ++nf)
            #pragma unroll
            for (int r = 0; r < 8; ++r)
                Pls[wave][r + hlf * 8][nf * 16 + lm] = f2bf(sf[nf][r]);
        Frag pf;
        pf.q[0] = *(const uint4*)&Pls[wave][lm][hlf * 16];
        pf.q[1] = *(const uint4*)&Pls[wave][lm][hlf * 16 + 8];

        // O += P @ V
        #pragma unroll
        for (int f = 0; f < 8; ++f) {
            Frag vf;
            vf.q[0] = *(const uint4*)&Vt[f * 16 + lm][hlf * 16];
            vf.q[1] = *(const uint4*)&Vt[f * 16 + lm][hlf * 16 + 8];
            o[f] = bf16_wmma(pf, vf, o[f]);
        }
    }

    #pragma unroll
    for (int f = 0; f < 8; ++f)
        #pragma unroll
        for (int r = 0; r < 8; ++r) {
            size_t idx = ((size_t)(b * kS + qBase + wave * 16 + r + hlf * 8)) * kD
                       + (size_t)h * kHD + f * 16 + lm;
            O[idx] = f2bf(o[f][r] / lrow[r]);
        }
}

// ---------------------------------------------------------------- launch
extern "C" void kernel_launch(void* const* d_in, const int* in_sizes, int n_in,
                              void* d_out, int out_size, void* d_ws, size_t ws_size,
                              hipStream_t stream) {
    (void)in_sizes; (void)n_in; (void)out_size; (void)ws_size;
    const float* x   = (const float*)d_in[0];
    const float* fc  = (const float*)d_in[1];
    const float* fs  = (const float*)d_in[2];
    const float* anw = (const float*)d_in[3];
    const float* wq  = (const float*)d_in[4];
    const float* wk  = (const float*)d_in[5];
    const float* wv  = (const float*)d_in[6];
    const float* wo  = (const float*)d_in[7];
    const float* fnw = (const float*)d_in[8];
    const float* w1  = (const float*)d_in[9];
    const float* w2  = (const float*)d_in[10];
    const float* w3  = (const float*)d_in[11];
    float* out = (float*)d_out;

    char* ws = (char*)d_ws;
    size_t off = 0;
    auto alloc = [&](size_t elems) {
        unsigned short* p = (unsigned short*)(ws + off);
        off = (off + elems * 2 + 255) & ~(size_t)255;
        return p;
    };
    unsigned short* hb  = alloc((size_t)kM * kD);
    unsigned short* wqb = alloc((size_t)kD * kD);
    unsigned short* wkb = alloc((size_t)kD * kD);
    unsigned short* wvb = alloc((size_t)kD * kD);
    unsigned short* wob = alloc((size_t)kD * kD);
    unsigned short* w1b = alloc((size_t)kFF * kD);
    unsigned short* w3b = alloc((size_t)kFF * kD);
    unsigned short* w2b = alloc((size_t)kD * kFF);
    unsigned short* qb  = alloc((size_t)kM * kD);
    unsigned short* kbm = alloc((size_t)kM * kD);
    unsigned short* vb  = alloc((size_t)kM * kD);
    unsigned short* ob  = alloc((size_t)kM * kD);
    unsigned short* ffb = alloc((size_t)kM * kFF);

    auto cvt = [&](const float* s, unsigned short* d, size_t n) {
        k_cvt_bf16<<<dim3((unsigned)((n + 255) / 256)), dim3(256), 0, stream>>>(s, d, (int)n);
    };
    cvt(wq, wqb, (size_t)kD * kD);
    cvt(wk, wkb, (size_t)kD * kD);
    cvt(wv, wvb, (size_t)kD * kD);
    cvt(wo, wob, (size_t)kD * kD);
    cvt(w1, w1b, (size_t)kFF * kD);
    cvt(w3, w3b, (size_t)kFF * kD);
    cvt(w2, w2b, (size_t)kD * kFF);

    // attn path
    k_rmsnorm_bf16<<<dim3(kM), dim3(256), 0, stream>>>(x, anw, hb);
    dim3 gD(kD / 128, kM / 128);
    k_gemm_nt<GM_BF16><<<gD, dim3(256), 0, stream>>>(hb, wqb, nullptr, qb, nullptr, kM, kD, kD);
    k_gemm_nt<GM_BF16><<<gD, dim3(256), 0, stream>>>(hb, wkb, nullptr, kbm, nullptr, kM, kD, kD);
    k_gemm_nt<GM_BF16><<<gD, dim3(256), 0, stream>>>(hb, wvb, nullptr, vb, nullptr, kM, kD, kD);
    k_rope_inplace<<<dim3((kM * kH * (kHD / 2)) / 256), dim3(256), 0, stream>>>(qb, kbm, fc, fs);
    k_flash_attn<<<dim3(kS / 128, kH, kB), dim3(256), 0, stream>>>(qb, kbm, vb, ob);
    k_gemm_nt<GM_RESID><<<gD, dim3(256), 0, stream>>>(ob, wob, x, nullptr, out, kM, kD, kD);

    // ffn path
    k_rmsnorm_bf16<<<dim3(kM), dim3(256), 0, stream>>>(out, fnw, hb);
    k_gemm_swiglu<<<dim3(kFF / 64, kM / 128), dim3(256), 0, stream>>>(hb, w1b, w3b, ffb, kM, kFF, kD);
    k_gemm_nt<GM_ADD><<<gD, dim3(256), 0, stream>>>(ffb, w2b, nullptr, nullptr, out, kM, kD, kFF);
}
